// Indexer_73040213835928
// MI455X (gfx1250) — 8-bit matrix operand order repaired and checked by reading, compiled, not run on hardware
//
#include <hip/hip_runtime.h>
#include <hip/hip_bf16.h>
#include <math.h>


typedef int   v16i __attribute__((ext_vector_type(16)));
typedef float v8f  __attribute__((ext_vector_type(8)));

#define HEADS     64
#define HDIM      128
#define TOPK_K    2048
#define TMAX      4096
#define NEG_INF   (-__builtin_inff())
#define FLT_MAX_C 3.402823466e38f

__device__ inline unsigned int f32_to_e4m3(float x) {
    unsigned int ub   = __float_as_uint(x);
    unsigned int sign = (ub >> 31) << 7;
    float a = __uint_as_float(ub & 0x7FFFFFFFu);
    if (a < 0.015625f) {
        unsigned int q = (unsigned int)__float2int_rn(a * 512.0f);
        return sign | q;
    }
    unsigned int b   = __float_as_uint(a);
    unsigned int lsb = (b >> 20) & 1u;
    b += 0x000FFFFFu + lsb;
    int          E = (int)(b >> 23) - 120;
    unsigned int m = (b >> 20) & 7u;
    if (E > 15 || (E == 15 && m > 6)) return sign | 0x7Eu;
    return sign | ((unsigned int)E << 3) | m;
}

__device__ inline void fwht128(float& x0, float& x1, float& x2, float& x3, int lane) {
    { float t = x0; x0 = t + x1; x1 = t - x1; }
    { float t = x2; x2 = t + x3; x3 = t - x3; }
    { float t = x0; x0 = t + x2; x2 = t - x2; }
    { float t = x1; x1 = t + x3; x3 = t - x3; }
#pragma unroll
    for (int m = 1; m <= 16; m <<= 1) {
        float y0 = __shfl_xor(x0, m, 32);
        float y1 = __shfl_xor(x1, m, 32);
        float y2 = __shfl_xor(x2, m, 32);
        float y3 = __shfl_xor(x3, m, 32);
        bool up = (lane & m) != 0;
        x0 = up ? (y0 - x0) : (x0 + y0);
        x1 = up ? (y1 - x1) : (x1 + y1);
        x2 = up ? (y2 - x2) : (x2 + y2);
        x3 = up ? (y3 - x3) : (x3 + y3);
    }
}

__global__ void bounds_kernel(const int* __restrict__ seq_lens, int nseq,
                              int* __restrict__ cu_ks, int T) {
    int t = blockIdx.x * blockDim.x + threadIdx.x;
    if (t >= T) return;
    int start = 0, acc = 0;
    for (int i = 0; i < nseq; ++i) {
        int next = acc + seq_lens[i];
        if (t >= acc && t < next) start = acc;
        acc = next;
    }
    cu_ks[t] = start;
}

__global__ __launch_bounds__(256) void rotq_q_kernel(
    const float* __restrict__ q, const float* __restrict__ weights,
    unsigned char* __restrict__ qf, float* __restrict__ w_out, int nrows) {
    int wid  = (int)((blockIdx.x * blockDim.x + threadIdx.x) >> 5);
    int lane = threadIdx.x & 31;
    if (wid >= nrows) return;

    const float4 v = *(const float4*)(q + (size_t)wid * HDIM + 4 * lane);
    float x0 = v.x, x1 = v.y, x2 = v.z, x3 = v.w;
    fwht128(x0, x1, x2, x3, lane);
    const float hs = 0.08838834764831845f;
    x0 *= hs; x1 *= hs; x2 *= hs; x3 *= hs;

    float amax = fmaxf(fmaxf(fabsf(x0), fabsf(x1)), fmaxf(fabsf(x2), fabsf(x3)));
#pragma unroll
    for (int m = 16; m; m >>= 1) amax = fmaxf(amax, __shfl_xor(amax, m, 32));
    float scale = fmaxf(amax, 1e-4f) * (1.0f / 448.0f);
    float inv   = 1.0f / scale;

    unsigned int word = f32_to_e4m3(x0 * inv)
                      | (f32_to_e4m3(x1 * inv) << 8)
                      | (f32_to_e4m3(x2 * inv) << 16)
                      | (f32_to_e4m3(x3 * inv) << 24);
    *(unsigned int*)(qf + (size_t)wid * HDIM + 4 * lane) = word;

    if (lane == 0) w_out[wid] = weights[wid] * scale * hs;
}

__global__ __launch_bounds__(256) void rotq_k_kernel(
    const float* __restrict__ k, unsigned char* __restrict__ kf,
    float* __restrict__ k_scale, int T) {
    int wid  = (int)((blockIdx.x * blockDim.x + threadIdx.x) >> 5);
    int lane = threadIdx.x & 31;
    if (wid >= T) return;

    const float4 v = *(const float4*)(k + (size_t)wid * HDIM + 4 * lane);
    float x0 = v.x, x1 = v.y, x2 = v.z, x3 = v.w;
    fwht128(x0, x1, x2, x3, lane);
    const float hs = 0.08838834764831845f;
    x0 *= hs; x1 *= hs; x2 *= hs; x3 *= hs;

    float amax = fmaxf(fmaxf(fabsf(x0), fabsf(x1)), fmaxf(fabsf(x2), fabsf(x3)));
#pragma unroll
    for (int m = 16; m; m >>= 1) amax = fmaxf(amax, __shfl_xor(amax, m, 32));
    float scale = fmaxf(amax, 1e-4f) * (1.0f / 448.0f);
    float inv   = 1.0f / scale;

    unsigned int word = f32_to_e4m3(x0 * inv)
                      | (f32_to_e4m3(x1 * inv) << 8)
                      | (f32_to_e4m3(x2 * inv) << 16)
                      | (f32_to_e4m3(x3 * inv) << 24);
    *(unsigned int*)(kf + (size_t)wid * HDIM + 4 * lane) = word;

    if (lane == 0) k_scale[wid] = scale;
}

__device__ inline v16i load_afrag(const unsigned char* qrow, int hi) {
    v16i a;
#pragma unroll
    for (int r = 0; r < 16; ++r) {
        int rr = r & 7;
        int kd = (r >> 3) * 64 + ((rr >> 1) * 4 + (rr & 1) + hi * 2) * 4;
        a[r] = *(const int*)(qrow + kd);
    }
    return a;
}

__device__ inline v16i load_bfrag(const unsigned char* krow, int hi) {
    v16i b;
#pragma unroll
    for (int r = 0; r < 16; ++r) {
        int kbase = (r >> 3) * 64 + (((r & 7) >> 1) * 4 + (r & 1) + hi * 2) * 4;
        b[r] = *(const int*)(krow + kbase);
    }
    return b;
}

__device__ inline float relu_f(float x) {
    return __builtin_amdgcn_fmed3f(x, 0.0f, FLT_MAX_C);
}

__global__ __launch_bounds__(32) void logits_kernel(
    const unsigned char* __restrict__ qf, const unsigned char* __restrict__ kf,
    const float* __restrict__ w, const float* __restrict__ k_scale,
    const int* __restrict__ cu_ks, float* __restrict__ out, int T) {

    const int t0    = blockIdx.y * 16;
    const int sbase = blockIdx.x * 64;
    const int lane  = threadIdx.x;
    const int hi    = lane >> 4;
    const int ln    = lane & 15;

    const int ks0 = cu_ks[t0];
    if (sbase > t0 + 15 || (sbase + 63) < ks0) {
#pragma unroll
        for (int j = 0; j < 4; ++j)
#pragma unroll
            for (int g = 0; g < 8; ++g) {
                int t = t0 + hi * 8 + g;
                out[(size_t)t * T + sbase + j * 16 + ln] = NEG_INF;
            }
        return;
    }

    __shared__ float wlds[HEADS * 16];
    for (int i = lane; i < HEADS * 16; i += 32)
        wlds[i] = w[(size_t)(t0 + (i & 15)) * HEADS + (i >> 4)];
    __syncthreads();

    const v16i b0 = load_bfrag(kf + (size_t)(sbase + 0  + ln) * HDIM, hi);
    const v16i b1 = load_bfrag(kf + (size_t)(sbase + 16 + ln) * HDIM, hi);
    const v16i b2 = load_bfrag(kf + (size_t)(sbase + 32 + ln) * HDIM, hi);
    const v16i b3 = load_bfrag(kf + (size_t)(sbase + 48 + ln) * HDIM, hi);

    v8f acc0 = {}, acc1 = {}, acc2 = {}, acc3 = {};

    const unsigned char* qrow = qf + (size_t)(t0 + ln) * HEADS * HDIM;
    v16i a_cur = load_afrag(qrow, hi);

    for (int h = 0; h < HEADS; ++h) {
        v16i a_nxt;
        if (h + 1 < HEADS)
            a_nxt = load_afrag(qrow + (h + 1) * HDIM, hi);

        v8f c = {};
        v8f s0 = __builtin_amdgcn_wmma_f32_16x16x128_fp8_fp8(a_cur, b0, (short)0, c, false, false);
        v8f s1 = __builtin_amdgcn_wmma_f32_16x16x128_fp8_fp8(a_cur, b1, (short)0, c, false, false);
        v8f s2 = __builtin_amdgcn_wmma_f32_16x16x128_fp8_fp8(a_cur, b2, (short)0, c, false, false);
        v8f s3 = __builtin_amdgcn_wmma_f32_16x16x128_fp8_fp8(a_cur, b3, (short)0, c, false, false);

        const float* wp = wlds + h * 16 + hi * 8;
#pragma unroll
        for (int g = 0; g < 8; ++g) {
            float wg = wp[g];
            acc0[g] += relu_f(s0[g]) * wg;
            acc1[g] += relu_f(s1[g]) * wg;
            acc2[g] += relu_f(s2[g]) * wg;
            acc3[g] += relu_f(s3[g]) * wg;
        }
        a_cur = a_nxt;
    }

#pragma unroll
    for (int j = 0; j < 4; ++j) {
        const int   s   = sbase + j * 16 + ln;
        const float ksc = k_scale[s];
        const v8f&  a   = (j == 0) ? acc0 : (j == 1) ? acc1 : (j == 2) ? acc2 : acc3;
#pragma unroll
        for (int g = 0; g < 8; ++g) {
            int  t     = t0 + hi * 8 + g;
            bool valid = (s >= cu_ks[t]) && (s <= t);
            out[(size_t)t * T + s] = valid ? a[g] * ksc : NEG_INF;
        }
    }
}

__global__ __launch_bounds__(256) void topk_kernel(
    const float* __restrict__ masked, int* __restrict__ idx_out, int T, int K) {
    __shared__ unsigned long long keys[TMAX];
    const int row = blockIdx.x;
    const float* src = masked + (size_t)row * T;

    for (int i = threadIdx.x; i < T; i += 256) {
        unsigned int u = __float_as_uint(src[i]);
        u = (u & 0x80000000u) ? ~u : (u | 0x80000000u);
        keys[i] = ((unsigned long long)(~u) << 32) | (unsigned int)i;
    }
    __syncthreads();

    for (int kk = 2; kk <= T; kk <<= 1) {
        for (int j = kk >> 1; j > 0; j >>= 1) {
            for (int i = threadIdx.x; i < T; i += 256) {
                int ixj = i ^ j;
                if (ixj > i) {
                    unsigned long long ka = keys[i], kb = keys[ixj];
                    bool asc  = ((i & kk) == 0);
                    bool swap = asc ? (ka > kb) : (ka < kb);
                    if (swap) { keys[i] = kb; keys[ixj] = ka; }
                }
            }
            __syncthreads();
        }
    }
    for (int i = threadIdx.x; i < K; i += 256)
        idx_out[(size_t)row * K + i] = (int)(keys[i] & 0xFFFFFFFFu);
}

extern "C" void kernel_launch(void* const* d_in, const int* in_sizes, int n_in,
                              void* d_out, int out_size, void* d_ws, size_t ws_size,
                              hipStream_t stream) {
    (void)n_in; (void)out_size; (void)ws_size;

    const int T    = in_sizes[0] / (HEADS * HDIM);
    const int nseq = in_sizes[3];

    const float* q        = (const float*)d_in[0];
    const float* k        = (const float*)d_in[1];
    const float* weights  = (const float*)d_in[2];
    const int*   seq_lens = (const int*)d_in[3];

    float* masked = (float*)d_out;
    int*   topk   = (int*)((char*)d_out + (size_t)T * T * sizeof(float));

    char* ws = (char*)d_ws;
    unsigned char* qf = (unsigned char*)ws;  ws += (size_t)T * HEADS * HDIM;
    unsigned char* kf = (unsigned char*)ws;  ws += (size_t)T * HDIM;
    float* w          = (float*)ws;          ws += (size_t)T * HEADS * sizeof(float);
    float* kscale     = (float*)ws;          ws += (size_t)T * sizeof(float);
    int*   cuks       = (int*)ws;

    bounds_kernel<<<(T + 255) / 256, 256, 0, stream>>>(seq_lens, nseq, cuks, T);

    rotq_q_kernel<<<(T * HEADS) / 8, 256, 0, stream>>>(q, weights, qf, w, T * HEADS);
    rotq_k_kernel<<<(T + 7) / 8, 256, 0, stream>>>(k, kf, kscale, T);

    dim3 grid(T / 64, T / 16);
    logits_kernel<<<grid, 32, 0, stream>>>(qf, kf, w, kscale, cuks, masked, T);

    topk_kernel<<<T, 256, 0, stream>>>(masked, topk, T, TOPK_K);
}
